// PPNet_49898930045408
// MI455X (gfx1250) — matrix operand order repaired and checked by reading, not compiled, not run on hardware
//
#include <hip/hip_runtime.h>
#include <hip/hip_bf16.h>
#include <cstdint>

// ---------------------------------------------------------------------------
// Types / WMMA helpers (gfx1250: wave32, v_wmma_f32_16x16x32_bf16)
// ---------------------------------------------------------------------------
typedef __attribute__((ext_vector_type(16))) __bf16 bf16x16;
typedef __attribute__((ext_vector_type(8)))  float  f32x8;

union FragB { uint32_t u[8]; bf16x16 v; };

__device__ __forceinline__ uint16_t f2b(float f) {
  __bf16 h = (__bf16)f;
  return __builtin_bit_cast(uint16_t, h);
}
__device__ __forceinline__ float b2f(uint16_t u) {
  return (float)__builtin_bit_cast(__bf16, u);
}

__device__ __forceinline__ f32x8 wmma_bf16(const FragB& a, const FragB& b, f32x8 c) {
  // D = A(16x32,bf16) * B(32x16,bf16) + C(16x16,f32)
  return __builtin_amdgcn_wmma_f32_16x16x32_bf16(
      /*neg_a=*/false, a.v, /*neg_b=*/false, b.v,
      /*c_mod=*/(short)0, c, /*reuse_a=*/false, /*reuse_b=*/false);
}

// A fragment: 16x32 bf16, row-major, row stride ld (elements). lane = M row.
// VGPR v: K = (v/4)*16 + (lane/16)*8 + (v%4)*2  (pairs contiguous in K).
__device__ __forceinline__ FragB load_a(const uint16_t* A, int ld) {
  int lane = threadIdx.x & 31;
  int m = lane & 15, h = lane >> 4;
  const uint16_t* row = A + (long)m * ld + h * 8;
  FragB f;
#pragma unroll
  for (int v = 0; v < 8; ++v) {
    int k = ((v >> 2) << 4) + ((v & 3) << 1);
    f.u[v] = *(const uint32_t*)(row + k);
  }
  return f;
}

// B fragment: 32x16 bf16 (KxN), stored column-major (each output column is a
// contiguous K-run of length ld). lane%16 = column N, lane/16 selects K offset 8 (same K order as load_a).
__device__ __forceinline__ FragB load_b(const uint16_t* B, int ld) {
  int lane = threadIdx.x & 31;
  int n = lane & 15, h = lane >> 4;
  const uint16_t* col = B + (long)n * ld + h * 8;
  FragB f;
#pragma unroll
  for (int v = 0; v < 8; ++v) f.u[v] = *(const uint32_t*)(col + ((v >> 2) << 4) + ((v & 3) << 1));
  return f;
}

// ---------------------------------------------------------------------------
// Generic WMMA GEMM:  out = [relu]( X[MxKp](bf16) * Wt + bias + pe ) + resid
// Wt stored transposed (Nout x Kp, bf16).
// One wave computes a 16x64 output strip: one A fragment feeds 4 independent
// accumulators (4x A reuse, no WMMA->WMMA RAW on the same destination).
// ---------------------------------------------------------------------------
__global__ void k_gemm(const uint16_t* __restrict__ X, int ldx,
                       const uint16_t* __restrict__ Wt, int Kp,
                       const float* __restrict__ bias,
                       const float* __restrict__ resid,
                       const float* __restrict__ pe,
                       float* __restrict__ outf, uint16_t* __restrict__ outb,
                       int M, int Nout, int relu) {
  int wid = threadIdx.x >> 5;
  int nstr = Nout >> 6;                       // strips of 64 columns
  long tile = (long)blockIdx.x * 8 + wid;
  long tm = tile / nstr;
  int  ts = (int)(tile % nstr);
  if (tm >= (M >> 4)) return;
  const uint16_t* Arow = X + tm * 16 * (long)ldx;
  const uint16_t* B0 = Wt + (long)(ts * 64) * Kp;
  f32x8 cc[4];
#pragma unroll
  for (int t = 0; t < 4; ++t) cc[t] = (f32x8){0.f,0.f,0.f,0.f,0.f,0.f,0.f,0.f};
  for (int k = 0; k < Kp; k += 32) {
    FragB a = load_a(Arow + k, ldx);
#pragma unroll
    for (int t = 0; t < 4; ++t) {
      FragB b = load_b(B0 + (long)(t * 16) * Kp + k, Kp);
      cc[t] = wmma_bf16(a, b, cc[t]);
    }
  }
  int lane = threadIdx.x & 31;
  long rbase = tm * 16 + ((lane >> 4) << 3);
#pragma unroll
  for (int t = 0; t < 4; ++t) {
    int cn = ts * 64 + t * 16 + (lane & 15);
#pragma unroll
    for (int v = 0; v < 8; ++v) {
      long rm = rbase + v;
      float val = cc[t][v];
      if (bias) val += bias[cn];
      if (pe)   val += pe[(rm & 63) * 256 + cn];   // positional encoding, t = row%64
      if (relu) val = fmaxf(val, 0.f);
      long idx = rm * (long)Nout + cn;
      if (resid) val += resid[idx];
      if (outf) outf[idx] = val;
      if (outb) outb[idx] = f2b(val);
    }
  }
}

// ---------------------------------------------------------------------------
// Weight convert+transpose: Wt[n*Kpad+k] = bf16(W[k*Nout+n]), zero pad K.
// ---------------------------------------------------------------------------
__global__ void k_w2bf(const float* __restrict__ W, uint16_t* __restrict__ Wt,
                       int K, int Nout, int Kpad) {
  long idx = (long)blockIdx.x * 256 + threadIdx.x;
  if (idx >= (long)Nout * Kpad) return;
  int n = (int)(idx / Kpad), k = (int)(idx % Kpad);
  float v = (k < K) ? W[(long)k * Nout + n] : 0.f;
  Wt[idx] = f2b(v);
}

// ---------------------------------------------------------------------------
// Gate: softmax over 158 logits of mkt@gate_W + b, scaled by D_FEAT.
// ---------------------------------------------------------------------------
__global__ void k_gate(const float* __restrict__ mkt, const float* __restrict__ W,
                       const float* __restrict__ b, float* __restrict__ gate) {
  __shared__ float sg[160];
  int n = blockIdx.x, tid = threadIdx.x;
  if (tid < 158) {
    const float* m = mkt + ((long)n * 64 + 63) * 64;
    float a = b[tid];
    for (int j = 0; j < 64; ++j) a += m[j] * W[j * 158 + tid];
    sg[tid] = a;
  }
  __syncthreads();
  float mx = -1e30f;
  for (int j = 0; j < 158; ++j) mx = fmaxf(mx, sg[j]);
  float sum = 0.f;
  for (int j = 0; j < 158; ++j) sum += __expf(sg[j] - mx);
  if (tid < 158) gate[(long)n * 158 + tid] = 158.0f * __expf(sg[tid] - mx) / sum;
}

// Xg[nt,f] = bf16(stock[nt,f]*gate[n,f]) for f<158, zero pad to 160
__global__ void k_scale(const float* __restrict__ sf, const float* __restrict__ gate,
                        uint16_t* __restrict__ Xg) {
  long idx = (long)blockIdx.x * 256 + threadIdx.x;
  const long total = 4096L * 64 * 160;
  if (idx >= total) return;
  int f = (int)(idx % 160);
  long nt = idx / 160;
  long n = nt >> 6;
  float v = (f < 158) ? sf[nt * 158 + f] * gate[n * 158 + f] : 0.f;
  Xg[idx] = f2b(v);
}

// ---------------------------------------------------------------------------
// LayerNorm over D=256; optional f32 out (stride 256) and bf16 out (stride ldo)
// ---------------------------------------------------------------------------
__global__ void k_ln(const float* __restrict__ x, const float* __restrict__ g,
                     const float* __restrict__ b, float* __restrict__ outf,
                     uint16_t* __restrict__ outb, int ldo) {
  __shared__ float red[256];
  int row = blockIdx.x, tid = threadIdx.x;
  float v = x[(long)row * 256 + tid];
  red[tid] = v; __syncthreads();
  for (int s = 128; s > 0; s >>= 1) { if (tid < s) red[tid] += red[tid + s]; __syncthreads(); }
  float mean = red[0] * (1.f / 256.f); __syncthreads();
  float d = v - mean;
  red[tid] = d * d; __syncthreads();
  for (int s = 128; s > 0; s >>= 1) { if (tid < s) red[tid] += red[tid + s]; __syncthreads(); }
  float inv = rsqrtf(red[0] * (1.f / 256.f) + 1e-5f);
  float o = d * inv * g[tid] + b[tid];
  if (outf) outf[(long)row * 256 + tid] = o;
  if (outb) outb[(long)row * ldo + tid] = f2b(o);
}

// LN over gathered embedding rows (D=32) -> bf16 into qkin cols 256..287
__global__ void k_ln_emb(const int* __restrict__ ids, const float* __restrict__ emb,
                         const float* __restrict__ g, const float* __restrict__ b,
                         uint16_t* __restrict__ outb) {
  __shared__ float red[32];
  int row = blockIdx.x, tid = threadIdx.x;
  const float* e = emb + (long)ids[row] * 32;
  float v = e[tid];
  red[tid] = v; __syncthreads();
  for (int s = 16; s > 0; s >>= 1) { if (tid < s) red[tid] += red[tid + s]; __syncthreads(); }
  float mean = red[0] * (1.f / 32.f); __syncthreads();
  float d = v - mean;
  red[tid] = d * d; __syncthreads();
  for (int s = 16; s > 0; s >>= 1) { if (tid < s) red[tid] += red[tid + s]; __syncthreads(); }
  float inv = rsqrtf(red[0] * (1.f / 32.f) + 1e-5f);
  outb[(long)row * 288 + tid] = f2b(d * inv * g[tid] + b[tid]);
}

// ---------------------------------------------------------------------------
// Temporal attention: one block per (stock, head). S=QK^T (no scale, per ref),
// softmax over s, O=P@V accumulated into residual X. All contractions WMMA.
// ---------------------------------------------------------------------------
__global__ void k_tattn(const uint16_t* __restrict__ Q, const uint16_t* __restrict__ K,
                        const uint16_t* __restrict__ V, float* __restrict__ X) {
  __shared__ __align__(16) float    S[64 * 64];
  __shared__ __align__(16) uint16_t P[64 * 64];
  __shared__ __align__(16) uint16_t Vt[64 * 64];
  int n = blockIdx.x, h = blockIdx.y;
  int tid = threadIdx.x, wid = tid >> 5, lane = tid & 31;
  long base = ((long)n * 64) * 256 + h * 64;
  const uint16_t* Qp = Q + base;
  const uint16_t* Kp = K + base;
  const uint16_t* Vp = V + base;
  for (int i = tid; i < 4096; i += 256) {       // V -> LDS transposed (d-major)
    int d = i >> 6, s = i & 63;
    Vt[d * 64 + s] = Vp[(long)s * 256 + d];
  }
  for (int tt = wid; tt < 16; tt += 8) {        // S = Q K^T
    int mt = tt >> 2, nt = tt & 3;
    f32x8 c = {0.f, 0.f, 0.f, 0.f, 0.f, 0.f, 0.f, 0.f};
#pragma unroll
    for (int k = 0; k < 64; k += 32) {
      FragB a = load_a(Qp + (long)mt * 16 * 256 + k, 256);
      FragB b = load_b(Kp + (long)nt * 16 * 256 + k, 256);
      c = wmma_bf16(a, b, c);
    }
#pragma unroll
    for (int v = 0; v < 8; ++v)
      S[(mt * 16 + ((lane >> 4) << 3) + v) * 64 + nt * 16 + (lane & 15)] = c[v];
  }
  __syncthreads();
  if (tid < 64) {                               // row softmax
    float mx = -1e30f;
    for (int j = 0; j < 64; ++j) mx = fmaxf(mx, S[tid * 64 + j]);
    float sum = 0.f;
    for (int j = 0; j < 64; ++j) { float e = __expf(S[tid * 64 + j] - mx); S[tid * 64 + j] = e; sum += e; }
    float inv = 1.f / sum;
    for (int j = 0; j < 64; ++j) P[tid * 64 + j] = f2b(S[tid * 64 + j] * inv);
  }
  __syncthreads();
  for (int tt = wid; tt < 16; tt += 8) {        // O = P V  -> X += O
    int mt = tt >> 2, nt = tt & 3;
    f32x8 c = {0.f, 0.f, 0.f, 0.f, 0.f, 0.f, 0.f, 0.f};
#pragma unroll
    for (int k = 0; k < 64; k += 32) {
      FragB a = load_a((const uint16_t*)P + mt * 16 * 64 + k, 64);
      FragB b = load_b((const uint16_t*)Vt + nt * 16 * 64 + k, 64);
      c = wmma_bf16(a, b, c);
    }
#pragma unroll
    for (int v = 0; v < 8; ++v) {
      int row = mt * 16 + ((lane >> 4) << 3) + v;
      int col = nt * 16 + (lane & 15);
      long idx = base + (long)row * 256 + col;
      X[idx] += c[v];
    }
  }
}

// ---------------------------------------------------------------------------
// lam/z aggregation: lam = softmax_t(h_t . h_63); z = sum_t lam_t x_t
// ---------------------------------------------------------------------------
__global__ void k_lamz(const uint16_t* __restrict__ H, const float* __restrict__ X,
                       float* __restrict__ Z) {
  __shared__ float lam[64];
  int n = blockIdx.x, tid = threadIdx.x;
  long rb = (long)n * 64;
  if (tid < 64) {
    const uint16_t* hr = H + (rb + tid) * 256;
    const uint16_t* hl = H + (rb + 63) * 256;
    float s = 0.f;
    for (int d = 0; d < 256; ++d) s += b2f(hr[d]) * b2f(hl[d]);
    lam[tid] = s;
  }
  __syncthreads();
  float mx = -1e30f;
  for (int j = 0; j < 64; ++j) mx = fmaxf(mx, lam[j]);
  float sum = 0.f;
  for (int j = 0; j < 64; ++j) sum += __expf(lam[j] - mx);
  __syncthreads();
  if (tid < 64) lam[tid] = __expf(lam[tid] - mx) / sum;
  __syncthreads();
  float acc = 0.f;
  const float* xr = X + rb * 256 + tid;
  for (int t = 0; t < 64; ++t) acc += lam[t] * xr[(long)t * 256];
  Z[(long)n * 256 + tid] = acc;
}

// copy zn bf16 into qkin (stride 288, cols 0..255)
__global__ void k_copy(const uint16_t* __restrict__ src, uint16_t* __restrict__ qkin) {
  long idx = (long)blockIdx.x * 256 + threadIdx.x;
  if (idx >= 4096L * 256) return;
  long r = idx >> 8; int c = (int)(idx & 255);
  qkin[r * 288 + c] = src[idx];
}

// V2 (4096x256) -> V2t (256x4096), key index contiguous per (head,d)
__global__ void k_trv2(const uint16_t* __restrict__ V2, uint16_t* __restrict__ V2t) {
  long idx = (long)blockIdx.x * 256 + threadIdx.x;
  if (idx >= 4096L * 256) return;
  long m = idx >> 8; int c = (int)(idx & 255);
  V2t[(long)c * 4096 + m] = V2[idx];
}

// ---------------------------------------------------------------------------
// Cross-sectional flash attention: one wave per (head, 16-query tile).
// Streams 4096 keys in chunks of 32 with online softmax; P via wave-private LDS.
// ---------------------------------------------------------------------------
__global__ void k_sattn(const uint16_t* __restrict__ Q2, const uint16_t* __restrict__ K2,
                        const uint16_t* __restrict__ V2t, const float* __restrict__ zn,
                        float* __restrict__ out) {
  __shared__ __align__(16) uint16_t P[8][16 * 32];
  int tid = threadIdx.x, wid = tid >> 5, lane = tid & 31, half = lane >> 4;
  int wg = blockIdx.x * 8 + wid;   // 0..1023
  int head = wg >> 8;              // 0..3
  int qt = wg & 255;               // 0..255
  const uint16_t* Qrow = Q2 + (long)qt * 16 * 256 + head * 64;
  FragB aQ0 = load_a(Qrow, 256);
  FragB aQ1 = load_a(Qrow + 32, 256);
  float mst[8], lst[8];
#pragma unroll
  for (int v = 0; v < 8; ++v) { mst[v] = -1e30f; lst[v] = 0.f; }
  f32x8 o0 = {0.f,0.f,0.f,0.f,0.f,0.f,0.f,0.f};
  f32x8 o1 = o0, o2 = o0, o3 = o0;
  const uint16_t* Kb = K2 + head * 64;
  const uint16_t* Vb = V2t + (long)head * 64 * 4096;
  uint16_t* Pw = &P[wid][0];
  for (int kc = 0; kc < 4096; kc += 32) {
    const uint16_t* K0 = Kb + (long)kc * 256;
    f32x8 s0 = {0.f,0.f,0.f,0.f,0.f,0.f,0.f,0.f};
    f32x8 s1 = s0;
    { FragB b = load_b(K0, 256);             s0 = wmma_bf16(aQ0, b, s0); }
    { FragB b = load_b(K0 + 32, 256);        s0 = wmma_bf16(aQ1, b, s0); }
    { FragB b = load_b(K0 + 16 * 256, 256);      s1 = wmma_bf16(aQ0, b, s1); }
    { FragB b = load_b(K0 + 16 * 256 + 32, 256); s1 = wmma_bf16(aQ1, b, s1); }
#pragma unroll
    for (int v = 0; v < 8; ++v) {
      float x0 = s0[v] * 0.125f, x1 = s1[v] * 0.125f;   // 1/sqrt(64)
      float r = fmaxf(x0, x1);
      for (int off = 8; off >= 1; off >>= 1) r = fmaxf(r, __shfl_xor(r, off, 16));
      float mnew = fmaxf(mst[v], r);
      float alpha = __expf(mst[v] - mnew);
      float p0 = __expf(x0 - mnew), p1 = __expf(x1 - mnew);
      float rs = p0 + p1;
      for (int off = 8; off >= 1; off >>= 1) rs += __shfl_xor(rs, off, 16);
      lst[v] = lst[v] * alpha + rs;
      mst[v] = mnew;
      int row = half * 8 + v;
      Pw[row * 32 + (lane & 15)]      = f2b(p0);
      Pw[row * 32 + 16 + (lane & 15)] = f2b(p1);
      o0[v] *= alpha; o1[v] *= alpha; o2[v] *= alpha; o3[v] *= alpha;
    }
    __syncthreads();
    FragB aP = load_a(Pw, 32);
    { FragB b = load_b(Vb + kc, 4096);              o0 = wmma_bf16(aP, b, o0); }
    { FragB b = load_b(Vb + 16L * 4096 + kc, 4096); o1 = wmma_bf16(aP, b, o1); }
    { FragB b = load_b(Vb + 32L * 4096 + kc, 4096); o2 = wmma_bf16(aP, b, o2); }
    { FragB b = load_b(Vb + 48L * 4096 + kc, 4096); o3 = wmma_bf16(aP, b, o3); }
    __syncthreads();
  }
#pragma unroll
  for (int v = 0; v < 8; ++v) {
    float inv = 1.f / lst[v];
    int row = qt * 16 + half * 8 + v;
    int c0 = head * 64 + (lane & 15);
    long b_ = (long)row * 256;
    out[b_ + c0]      = zn[b_ + c0]      + o0[v] * inv;   // xt2 = zn + o2
    out[b_ + c0 + 16] = zn[b_ + c0 + 16] + o1[v] * inv;
    out[b_ + c0 + 32] = zn[b_ + c0 + 32] + o2[v] * inv;
    out[b_ + c0 + 48] = zn[b_ + c0 + 48] + o3[v] * inv;
  }
}

// final decode: out[n] = ff[n,:] . dec_W + dec_b
__global__ void k_dec(const float* __restrict__ ff, const float* __restrict__ w,
                      const float* __restrict__ b, float* __restrict__ out) {
  __shared__ float red[256];
  int n = blockIdx.x, tid = threadIdx.x;
  red[tid] = ff[(long)n * 256 + tid] * w[tid];
  __syncthreads();
  for (int s = 128; s > 0; s >>= 1) { if (tid < s) red[tid] += red[tid + s]; __syncthreads(); }
  if (tid == 0) out[n] = red[0] + b[0];
}

// ---------------------------------------------------------------------------
// Host orchestration
// ---------------------------------------------------------------------------
extern "C" void kernel_launch(void* const* d_in, const int* in_sizes, int n_in,
                              void* d_out, int out_size, void* d_ws, size_t ws_size,
                              hipStream_t stream) {
  (void)in_sizes; (void)n_in; (void)out_size; (void)ws_size;
  const int*   ids   = (const int*)d_in[0];
  const float* sfeat = (const float*)d_in[1];
  const float* mfeat = (const float*)d_in[2];
  const float* gateW = (const float*)d_in[3];
  const float* gateb = (const float*)d_in[4];
  const float* embT  = (const float*)d_in[5];
  const float* projW = (const float*)d_in[6];
  const float* projb = (const float*)d_in[7];
  const float* pe    = (const float*)d_in[8];
  const float* taqW  = (const float*)d_in[9];
  const float* takW  = (const float*)d_in[10];
  const float* tavW  = (const float*)d_in[11];
  const float* tan1g = (const float*)d_in[12];
  const float* tan1b = (const float*)d_in[13];
  const float* tan2g = (const float*)d_in[14];
  const float* tan2b = (const float*)d_in[15];
  const float* taf1W = (const float*)d_in[16];
  const float* taf1b = (const float*)d_in[17];
  const float* taf2W = (const float*)d_in[18];
  const float* taf2b = (const float*)d_in[19];
  const float* aggW  = (const float*)d_in[20];
  const float* saqW  = (const float*)d_in[21];
  const float* sakW  = (const float*)d_in[22];
  const float* savW  = (const float*)d_in[23];
  const float* san1g = (const float*)d_in[24];
  const float* san1b = (const float*)d_in[25];
  const float* san2g = (const float*)d_in[26];
  const float* san2b = (const float*)d_in[27];
  const float* san3g = (const float*)d_in[28];
  const float* san3b = (const float*)d_in[29];
  const float* saf1W = (const float*)d_in[30];
  const float* saf1b = (const float*)d_in[31];
  const float* saf2W = (const float*)d_in[32];
  const float* saf2b = (const float*)d_in[33];
  const float* decW  = (const float*)d_in[34];
  const float* decb  = (const float*)d_in[35];
  float* out = (float*)d_out;

  const long NT = 4096L * 64;
  char* ws = (char*)d_ws;
  size_t off = 0;
  auto alloc = [&](size_t bytes) -> char* {
    char* p = ws + off; off += (bytes + 255) & ~(size_t)255; return p;
  };
  float*    X   = (float*)   alloc(NT * 256 * 4);  // f32 activations (x/xn/xt/x)
  uint16_t* B1  = (uint16_t*)alloc(NT * 256 * 2);  // bf16 staging
  uint16_t* Qb  = (uint16_t*)alloc(NT * 256 * 2);
  uint16_t* Kb  = (uint16_t*)alloc(NT * 256 * 2);
  uint16_t* Vb  = (uint16_t*)alloc(NT * 256 * 2);
  uint16_t* Xg  = Qb;                              // alias: gated features (NT x 160)
  float*    gate   = (float*)   alloc(4096L * 158 * 4);
  uint16_t* projWt = (uint16_t*)alloc(256L * 160 * 2);
  uint16_t* taqWt  = (uint16_t*)alloc(256L * 256 * 2);
  uint16_t* takWt  = (uint16_t*)alloc(256L * 256 * 2);
  uint16_t* tavWt  = (uint16_t*)alloc(256L * 256 * 2);
  uint16_t* taf1Wt = (uint16_t*)alloc(256L * 256 * 2);
  uint16_t* taf2Wt = (uint16_t*)alloc(256L * 256 * 2);
  uint16_t* aggWt  = (uint16_t*)alloc(256L * 256 * 2);
  uint16_t* savWt  = (uint16_t*)alloc(256L * 256 * 2);
  uint16_t* saf1Wt = (uint16_t*)alloc(256L * 256 * 2);
  uint16_t* saf2Wt = (uint16_t*)alloc(256L * 256 * 2);
  uint16_t* saqWt  = (uint16_t*)alloc(256L * 288 * 2);
  uint16_t* sakWt  = (uint16_t*)alloc(256L * 288 * 2);
  float*    Z    = (float*)   alloc(4096L * 256 * 4);
  float*    ZN   = (float*)   alloc(4096L * 256 * 4);
  uint16_t* ZNb  = (uint16_t*)alloc(4096L * 256 * 2);
  uint16_t* qkin = (uint16_t*)alloc(4096L * 288 * 2);
  uint16_t* Q2   = (uint16_t*)alloc(4096L * 256 * 2);
  uint16_t* K2   = (uint16_t*)alloc(4096L * 256 * 2);
  uint16_t* V2   = (uint16_t*)alloc(4096L * 256 * 2);
  uint16_t* V2t  = (uint16_t*)alloc(4096L * 256 * 2);
  float*    XT2  = (float*)   alloc(4096L * 256 * 4);
  uint16_t* Y3b  = (uint16_t*)alloc(4096L * 256 * 2);
  uint16_t* H2b  = (uint16_t*)alloc(4096L * 256 * 2);
  float*    FF   = (float*)   alloc(4096L * 256 * 4);

  hipStream_t s = stream;
  const dim3 B256(256);
  // GEMM grids: one wave per 16x64 strip, 8 waves per block
  const unsigned GBIG   = (unsigned)(((NT >> 4) * 4) / 8);   // 8192 blocks
  const unsigned GSMALL = (unsigned)(((4096 >> 4) * 4) / 8); // 128 blocks

  // weight prep (bf16, transposed, K-padded)
  k_w2bf<<<dim3(160), B256, 0, s>>>(projW, projWt, 158, 256, 160);
  k_w2bf<<<dim3(256), B256, 0, s>>>(taqW,  taqWt,  256, 256, 256);
  k_w2bf<<<dim3(256), B256, 0, s>>>(takW,  takWt,  256, 256, 256);
  k_w2bf<<<dim3(256), B256, 0, s>>>(tavW,  tavWt,  256, 256, 256);
  k_w2bf<<<dim3(256), B256, 0, s>>>(taf1W, taf1Wt, 256, 256, 256);
  k_w2bf<<<dim3(256), B256, 0, s>>>(taf2W, taf2Wt, 256, 256, 256);
  k_w2bf<<<dim3(256), B256, 0, s>>>(aggW,  aggWt,  256, 256, 256);
  k_w2bf<<<dim3(256), B256, 0, s>>>(savW,  savWt,  256, 256, 256);
  k_w2bf<<<dim3(256), B256, 0, s>>>(saf1W, saf1Wt, 256, 256, 256);
  k_w2bf<<<dim3(256), B256, 0, s>>>(saf2W, saf2Wt, 256, 256, 256);
  k_w2bf<<<dim3(288), B256, 0, s>>>(saqW,  saqWt,  288, 256, 288);
  k_w2bf<<<dim3(288), B256, 0, s>>>(sakW,  sakWt,  288, 256, 288);

  // gating + feature scaling
  k_gate<<<dim3(4096), B256, 0, s>>>(mfeat, gateW, gateb, gate);
  k_scale<<<dim3((unsigned)((NT * 160 + 255) / 256)), B256, 0, s>>>(sfeat, gate, Xg);

  // x = Xg @ proj_W + b + pe   (f32 into X)
  k_gemm<<<dim3(GBIG), B256, 0, s>>>(Xg, 160, projWt, 160, projb, nullptr, pe,
                                     X, nullptr, (int)NT, 256, 0);
  // xn = LN(x) -> X(f32 in place) + B1(bf16)
  k_ln<<<dim3((unsigned)NT), B256, 0, s>>>(X, tan1g, tan1b, X, B1, 256);
  // q,k,v
  k_gemm<<<dim3(GBIG), B256, 0, s>>>(B1, 256, taqWt, 256, nullptr, nullptr, nullptr,
                                     nullptr, Qb, (int)NT, 256, 0);
  k_gemm<<<dim3(GBIG), B256, 0, s>>>(B1, 256, takWt, 256, nullptr, nullptr, nullptr,
                                     nullptr, Kb, (int)NT, 256, 0);
  k_gemm<<<dim3(GBIG), B256, 0, s>>>(B1, 256, tavWt, 256, nullptr, nullptr, nullptr,
                                     nullptr, Vb, (int)NT, 256, 0);
  // temporal attention: X += attn_out  (X becomes xt)
  k_tattn<<<dim3(4096, 4), B256, 0, s>>>(Qb, Kb, Vb, X);
  // FFN: LN -> f1(relu) -> f2 + residual ; also emit x bf16 into B1
  k_ln<<<dim3((unsigned)NT), B256, 0, s>>>(X, tan2g, tan2b, nullptr, B1, 256);
  k_gemm<<<dim3(GBIG), B256, 0, s>>>(B1, 256, taf1Wt, 256, taf1b, nullptr, nullptr,
                                     nullptr, Qb, (int)NT, 256, 1);
  k_gemm<<<dim3(GBIG), B256, 0, s>>>(Qb, 256, taf2Wt, 256, taf2b, X, nullptr,
                                     X, B1, (int)NT, 256, 0);
  // h = x @ agg_W ; lam/z aggregation
  k_gemm<<<dim3(GBIG), B256, 0, s>>>(B1, 256, aggWt, 256, nullptr, nullptr, nullptr,
                                     nullptr, Kb, (int)NT, 256, 0);
  k_lamz<<<dim3(4096), B256, 0, s>>>(Kb, X, Z);

  // stage 2: zn, en, qk_in
  k_ln<<<dim3(4096), B256, 0, s>>>(Z, san1g, san1b, ZN, ZNb, 256);
  k_copy<<<dim3(4096), B256, 0, s>>>(ZNb, qkin);
  k_ln_emb<<<dim3(4096), dim3(32), 0, s>>>(ids, embT, san2g, san2b, qkin + 256);
  // q2, k2, v2
  k_gemm<<<dim3(GSMALL), B256, 0, s>>>(qkin, 288, saqWt, 288, nullptr, nullptr, nullptr,
                                       nullptr, Q2, 4096, 256, 0);
  k_gemm<<<dim3(GSMALL), B256, 0, s>>>(qkin, 288, sakWt, 288, nullptr, nullptr, nullptr,
                                       nullptr, K2, 4096, 256, 0);
  k_gemm<<<dim3(GSMALL), B256, 0, s>>>(ZNb, 256, savWt, 256, nullptr, nullptr, nullptr,
                                       nullptr, V2, 4096, 256, 0);
  k_trv2<<<dim3(4096), B256, 0, s>>>(V2, V2t);
  // cross-sectional flash attention -> XT2 = zn + o2
  k_sattn<<<dim3(128), B256, 0, s>>>(Q2, K2, V2t, ZN, XT2);
  // FFN2 + residual
  k_ln<<<dim3(4096), B256, 0, s>>>(XT2, san3g, san3b, nullptr, Y3b, 256);
  k_gemm<<<dim3(GSMALL), B256, 0, s>>>(Y3b, 256, saf1Wt, 256, saf1b, nullptr, nullptr,
                                       nullptr, H2b, 4096, 256, 1);
  k_gemm<<<dim3(GSMALL), B256, 0, s>>>(H2b, 256, saf2Wt, 256, saf2b, XT2, nullptr,
                                       FF, nullptr, 4096, 256, 0);
  // decode
  k_dec<<<dim3(4096), B256, 0, s>>>(FF, decW, decb, out);
}
